// OneClassPDFGNN_43834436223264
// MI455X (gfx1250) — matrix operand order repaired and checked by reading, not compiled, not run on hardware
//
#include <hip/hip_runtime.h>

// ---------------------------------------------------------------------------
// CDNA5 (gfx1250) GCN autoencoder pipeline.
// Dense GEMMs use v_wmma_f32_16x16x32_bf16 with fragment-swizzled LDS staging
// so every lane's A/B fragment is one contiguous 32B chunk (2x ds_load_b128).
// Graph scatter/gather uses wave-per-edge coalesced loads + fp32 atomics.
// ---------------------------------------------------------------------------

typedef __attribute__((ext_vector_type(16))) __bf16 v16bf;
typedef __attribute__((ext_vector_type(8)))  float  v8f;

__device__ __forceinline__ __bf16 f2bf(float f) {
  // round-to-nearest-even fp32 -> bf16
  unsigned u = __builtin_bit_cast(unsigned, f);
  unsigned r = (u + 0x7FFFu + ((u >> 16) & 1u)) >> 16;
  unsigned short s = (unsigned short)r;
  return __builtin_bit_cast(__bf16, s);
}

// ---------------------------------------------------------------------------
// WMMA GEMM: Out[nrows,NC] = X[nrows,KDIM] @ W[KDIM,NC]  (+bias, +relu)
// 256 threads = 8 waves; each wave owns one 16x16 D tile.
// Requires (ROWS/16)*(NC/16) == 8, KDIM % 32 == 0, nrows % ROWS == 0.
//
// LDS layouts are fragment-swizzled (32B chunks, one chunk per lane per
// 16x16x32 step), contiguous k order k = 16*hi + elem:
//   A: X[r][32*kb+kin] -> chunk (kb*ROWS + r)*2 + (kin>>4),
//      elem kin&15
//   B: W[32*kb+kin][n] -> chunk (kb*NC + n)*2 + (kin>>4), elem kin&15
// ---------------------------------------------------------------------------
template <int ROWS, int KDIM, int NC, bool RELU, bool BIAS>
__global__ __launch_bounds__(256) void gemm_bf16_wmma(
    const float* __restrict__ X, const float* __restrict__ W,
    const float* __restrict__ bias, float* __restrict__ Out, int nrows) {
  __shared__ __bf16 sW[KDIM * NC];   // chunked: [KDIM/32][NC][2][16]
  __shared__ __bf16 sX[ROWS * KDIM]; // chunked: [KDIM/32][ROWS][2][16]
  const int tid = threadIdx.x;
  const int row_base = blockIdx.x * ROWS;
  if (row_base >= nrows) return;

  // Stage + swizzle W
  for (int i = tid; i < KDIM * NC; i += 256) {
    const int k = i / NC, n = i % NC;
    const int kb = k >> 5, kin = k & 31;
    sW[((kb * NC + n) * 2 + (kin >> 4)) * 16 + (kin & 15)] = f2bf(W[i]);
  }
  // Stage + swizzle X tile
  for (int i = tid; i < ROWS * KDIM; i += 256) {
    const int r = i / KDIM, k = i % KDIM;
    const int kb = k >> 5, kin = k & 31;
    const int e = kin & 15;
    sX[((kb * ROWS + r) * 2 + (kin >> 4)) * 16 + e] =
        f2bf(X[(size_t)row_base * KDIM + i]);
  }
  __syncthreads();

  const int lane = tid & 31;
  const int wv   = tid >> 5;
  constexpr int CT = NC / 16;   // column tiles
  const int m0 = (wv / CT) * 16;
  const int n0 = (wv % CT) * 16;
  const int hi = lane >> 4;     // which 16-lane half
  const int l15 = lane & 15;

  const v16bf* sXv = (const v16bf*)sX;
  const v16bf* sWv = (const v16bf*)sW;
  const int aBase = (m0 + l15) * 2 + hi;
  const int bBase = (n0 + l15) * 2 + hi;

  v8f acc = {0.f, 0.f, 0.f, 0.f, 0.f, 0.f, 0.f, 0.f};

#pragma unroll
  for (int kb = 0; kb < KDIM / 32; ++kb) {
    v16bf a = sXv[kb * ROWS * 2 + aBase];
    v16bf b = sWv[kb * NC * 2 + bBase];
    acc = __builtin_amdgcn_wmma_f32_16x16x32_bf16(
        false, a, false, b, (short)0, acc, false, false);
  }

  // D tile store: VGPR r -> row m0 + r + hi*8, col n0 + l15.
  const int gn = n0 + l15;
  const int gm = row_base + m0 + hi * 8;
  const float bv = BIAS ? bias[gn] : 0.f;
#pragma unroll
  for (int r = 0; r < 8; ++r) {
    float v = acc[r] + bv;
    if (RELU) v = fmaxf(v, 0.f);
    Out[(size_t)(gm + r) * NC + gn] = v;
  }
}

// ---------------------------------------------------------------------------
// Degree / normalization
// ---------------------------------------------------------------------------
__global__ void deg_init_kernel(float* deg, int n) {
  int i = blockIdx.x * blockDim.x + threadIdx.x;
  if (i < n) deg[i] = 1.0f;  // self loop
}

__global__ void deg_count_kernel(const int* __restrict__ col, float* deg, int e) {
  int i = blockIdx.x * blockDim.x + threadIdx.x;
  if (i < e) atomicAdd(&deg[col[i]], 1.0f);
}

__global__ void dinv_kernel(float* deg, int n) {
  int i = blockIdx.x * blockDim.x + threadIdx.x;
  if (i < n) {
    float d = deg[i];
    deg[i] = d > 0.f ? rsqrtf(d) : 0.f;
  }
}

// ---------------------------------------------------------------------------
// agg[i,:] = dinv[i]^2 * h[i,:]   (self-loop contribution), C = 64
// ---------------------------------------------------------------------------
__global__ void self_loop_init_kernel(const float* __restrict__ dinv,
                                      const float* __restrict__ h,
                                      float* __restrict__ agg, int total) {
  int i = blockIdx.x * blockDim.x + threadIdx.x;
  if (i < total) {
    float d = dinv[i >> 6];
    agg[i] = d * d * h[i];
  }
}

// ---------------------------------------------------------------------------
// Wave-per-edge scatter: agg[col,:] += dinv[row]*dinv[col] * h[row,:], C = 64
// Lane l handles channels l and l+32 -> two fully coalesced 128B row segments.
// ---------------------------------------------------------------------------
__global__ __launch_bounds__(256) void edge_scatter_kernel(
    const int* __restrict__ row, const int* __restrict__ col,
    const float* __restrict__ dinv, const float* __restrict__ h,
    float* __restrict__ agg, int e) {
  int wid = (int)((blockIdx.x * blockDim.x + threadIdx.x) >> 5);
  int lane = threadIdx.x & 31;
  if (wid >= e) return;
  int r = row[wid];
  int c = col[wid];
  float nrm = dinv[r] * dinv[c];
  float v0 = h[(size_t)r * 64 + lane];
  float v1 = h[(size_t)r * 64 + 32 + lane];
  atomicAdd(&agg[(size_t)c * 64 + lane], nrm * v0);
  atomicAdd(&agg[(size_t)c * 64 + 32 + lane], nrm * v1);
}

// ---------------------------------------------------------------------------
// dst[i] = relu(agg[i] + bias[i % 64])
// ---------------------------------------------------------------------------
__global__ void bias_relu_kernel(const float* __restrict__ agg,
                                 const float* __restrict__ bias,
                                 float* __restrict__ dst, int total) {
  int i = blockIdx.x * blockDim.x + threadIdx.x;
  if (i < total) dst[i] = fmaxf(agg[i] + bias[i & 63], 0.f);
}

// ---------------------------------------------------------------------------
// Wave-per-edge dot product over 64 channels + shfl_xor tree reduce (wave32)
// ---------------------------------------------------------------------------
__global__ __launch_bounds__(256) void edge_score_kernel(
    const int* __restrict__ row, const int* __restrict__ col,
    const float* __restrict__ nr, float* __restrict__ scores, int e) {
  int wid = (int)((blockIdx.x * blockDim.x + threadIdx.x) >> 5);
  int lane = threadIdx.x & 31;
  if (wid >= e) return;
  size_t rb = (size_t)row[wid] * 64;
  size_t cb = (size_t)col[wid] * 64;
  float s = nr[rb + lane] * nr[cb + lane] +
            nr[rb + 32 + lane] * nr[cb + 32 + lane];
#pragma unroll
  for (int off = 16; off > 0; off >>= 1) s += __shfl_xor(s, off, 32);
  if (lane == 0) scores[wid] = s;
}

// ---------------------------------------------------------------------------
extern "C" void kernel_launch(void* const* d_in, const int* in_sizes, int n_in,
                              void* d_out, int out_size, void* d_ws,
                              size_t ws_size, hipStream_t stream) {
  const float* x   = (const float*)d_in[0];
  const int*   ei  = (const int*)d_in[1];
  const float* W1  = (const float*)d_in[2];
  const float* b1  = (const float*)d_in[3];
  const float* W2  = (const float*)d_in[4];
  const float* b2  = (const float*)d_in[5];
  const float* Wd1 = (const float*)d_in[6];
  const float* bd1 = (const float*)d_in[7];
  const float* Wd2 = (const float*)d_in[8];
  const float* bd2 = (const float*)d_in[9];

  const int N = in_sizes[0] / 128;  // 100000
  const int E = in_sizes[1] / 2;    // 3200000
  const int* row = ei;
  const int* col = ei + E;

  // Output layout: [reconstructed_x N*128 | edge_scores E | node_rep N*64]
  float* out_rec    = (float*)d_out;
  float* out_scores = out_rec + (size_t)N * 128;
  float* out_nr     = out_scores + (size_t)E;

  // Workspace: dinv[N], bufA[N*64], bufB[N*64], bufC[N*64]
  float* dinv = (float*)d_ws;
  float* bufA = dinv + N;                 // h_lin1 -> h1 -> rec
  float* bufB = bufA + (size_t)N * 64;    // agg (both layers)
  float* bufC = bufB + (size_t)N * 64;    // h_lin2

  const int TB = 256;
  const int nodeBlk = (N + TB - 1) / TB;
  const int edgeBlk = (E + TB - 1) / TB;
  const int nc64Blk = ((int)((size_t)N * 64 + TB - 1) / TB);
  const int waveEdgeBlk = (E + 7) / 8;  // 8 waves per block, 1 edge per wave

  // --- normalization: deg (with self loop) -> dinv ---
  deg_init_kernel<<<nodeBlk, TB, 0, stream>>>(dinv, N);
  deg_count_kernel<<<edgeBlk, TB, 0, stream>>>(col, dinv, E);
  dinv_kernel<<<nodeBlk, TB, 0, stream>>>(dinv, N);

  // --- layer 1: h_lin1 = x @ W1 ---
  gemm_bf16_wmma<32, 128, 64, false, false>
      <<<N / 32, TB, 0, stream>>>(x, W1, nullptr, bufA, N);
  self_loop_init_kernel<<<nc64Blk, TB, 0, stream>>>(dinv, bufA, bufB, N * 64);
  edge_scatter_kernel<<<waveEdgeBlk, TB, 0, stream>>>(row, col, dinv, bufA,
                                                      bufB, E);
  bias_relu_kernel<<<nc64Blk, TB, 0, stream>>>(bufB, b1, bufA, N * 64);  // h1

  // --- layer 2: h_lin2 = h1 @ W2 ---
  gemm_bf16_wmma<32, 64, 64, false, false>
      <<<N / 32, TB, 0, stream>>>(bufA, W2, nullptr, bufC, N);
  self_loop_init_kernel<<<nc64Blk, TB, 0, stream>>>(dinv, bufC, bufB, N * 64);
  edge_scatter_kernel<<<waveEdgeBlk, TB, 0, stream>>>(row, col, dinv, bufC,
                                                      bufB, E);
  bias_relu_kernel<<<nc64Blk, TB, 0, stream>>>(bufB, b2, out_nr, N * 64);

  // --- decoder: rec = relu(nr @ Wd1 + bd1); out = rec @ Wd2 + bd2 ---
  gemm_bf16_wmma<32, 64, 64, true, true>
      <<<N / 32, TB, 0, stream>>>(out_nr, Wd1, bd1, bufA, N);
  gemm_bf16_wmma<16, 64, 128, false, true>
      <<<N / 16, TB, 0, stream>>>(bufA, Wd2, bd2, out_rec, N);

  // --- edge scores ---
  edge_score_kernel<<<waveEdgeBlk, TB, 0, stream>>>(row, col, out_nr,
                                                    out_scores, E);
}
